// RNN_40888088658583
// MI455X (gfx1250) — compile-verified, matrix operand order checked by reading, not run on hardware
//
#include <hip/hip_runtime.h>
#include <hip/hip_bf16.h>
#include <math.h>

typedef __attribute__((ext_vector_type(16))) __bf16 v16bf;
typedef __attribute__((ext_vector_type(8)))  float  v8f;
typedef unsigned int u32x4 __attribute__((ext_vector_type(4)));
typedef int          i32x8 __attribute__((ext_vector_type(8)));
typedef int          i32x4 __attribute__((ext_vector_type(4)));

#define T_STEPS 8192
#define D_IN    2048
#define N_NODES 2048
#define N_OUT   2048

#define REC_WGS  64
#define REC_COLS 32   /* N_NODES / REC_WGS */

#define BM 128
#define BN 32
#define BK 32

// ---------------------------------------------------------------------------
// Tensor Data Mover: 2D tile (tile_w x tile_h elements, 2-byte data) from a
// row-major matrix with row pitch `pitch_elems` into contiguous LDS.
// D# layout per CDNA5 ISA (group0 128b, group1 256b).
// Issued by one wave; completion tracked on TENSORcnt.
// ---------------------------------------------------------------------------
__device__ __forceinline__ void tdm_load_2d_bf16(unsigned lds_addr,
                                                 const void* gaddr,
                                                 unsigned pitch_elems,
                                                 unsigned tile_w,
                                                 unsigned tile_h)
{
  unsigned long long ga = (unsigned long long)gaddr;
  u32x4 g0;
  g0[0] = 1u;                                   // count=1, user descriptor
  g0[1] = lds_addr;                             // LDS byte address
  g0[2] = (unsigned)ga;                         // global_addr[31:0]
  g0[3] = (unsigned)((ga >> 32) & 0x1FFFFFFu)   // global_addr[56:32]
        | (2u << 30);                           // type=2 ("image")
  i32x8 g1;
  g1[0] = (int)(1u << 16);                      // wg_mask=0, data_size=1 (2B)
  g1[1] = (int)((pitch_elems & 0xFFFFu) << 16); // tensor_dim0[15:0]
  g1[2] = (int)((pitch_elems >> 16) |           // tensor_dim0[31:16]
                ((tile_h & 0xFFFFu) << 16));    // tensor_dim1[15:0]
  g1[3] = (int)((tile_h >> 16) |                // tensor_dim1[31:16]
                (tile_w << 16));                // tile_dim0
  g1[4] = (int)tile_h;                          // tile_dim1 (tile_dim2=0)
  g1[5] = (int)pitch_elems;                     // tensor_dim0_stride[31:0]
  g1[6] = 0;                                    // stride[47:32], d1_stride lo
  g1[7] = 0;
  i32x4 z4 = {0, 0, 0, 0};
  i32x8 z8 = {0, 0, 0, 0, 0, 0, 0, 0};
  __builtin_amdgcn_tensor_load_to_lds(g0, g1, z4, z4, z8, 0);
}

__device__ __forceinline__ unsigned lds_off(const void* p) {
  return (unsigned)(unsigned long long)p;       // low 32b of flat = LDS byte addr
}

// ---------------------------------------------------------------------------
// fp32 -> bf16 cast (grid-stride)
// ---------------------------------------------------------------------------
__global__ void cast_f32_bf16(const float* __restrict__ in,
                              __bf16* __restrict__ out, int n) {
  int i = blockIdx.x * blockDim.x + threadIdx.x;
  int stride = gridDim.x * blockDim.x;
  for (; i < n; i += stride) out[i] = (__bf16)in[i];
}

// Zero the h ping-pong buffers and the per-step barrier counters.
__global__ void init_state(float* __restrict__ h, int* __restrict__ bar) {
  int i = blockIdx.x * blockDim.x + threadIdx.x;
  if (i < 2 * N_NODES) h[i] = 0.f;
  if (i < T_STEPS)     bar[i] = 0;
}

// ---------------------------------------------------------------------------
// One K-step of the per-wave WMMA work: build A fragment + two B fragments
// from LDS tiles (CDNA5 16-bit operand layouts) and issue two WMMAs.
// ---------------------------------------------------------------------------
__device__ __forceinline__ void mma_step(const __bf16* __restrict__ at,
                                         const __bf16* __restrict__ bt,
                                         int wave, int half, int l16,
                                         v8f& acc0, v8f& acc1)
{
  union Frag { v16bf v; unsigned int u[8]; };
  Frag fa, fb0, fb1;
  const int arow = wave * 16 + l16;
#pragma unroll
  for (int v = 0; v < 8; ++v) {
    // VGPR0-3 -> K 0..15, VGPR4-7 -> K 16..31; half-wave picks the 8-wide
    // K octet; consecutive K pairs packed lo/hi in each dword.
    int k = ((v & 4) << 2) + (half << 3) + ((v & 3) << 1);
    fa.u[v] = *(const unsigned int*)&at[arow * BK + k];
    unsigned lo0 = *(const unsigned short*)&bt[k * BN + l16];
    unsigned hi0 = *(const unsigned short*)&bt[(k + 1) * BN + l16];
    fb0.u[v] = lo0 | (hi0 << 16);
    unsigned lo1 = *(const unsigned short*)&bt[k * BN + 16 + l16];
    unsigned hi1 = *(const unsigned short*)&bt[(k + 1) * BN + 16 + l16];
    fb1.u[v] = lo1 | (hi1 << 16);
  }
  acc0 = __builtin_amdgcn_wmma_f32_16x16x32_bf16(
      false, fa.v, false, fb0.v, (short)0, acc0, false, false);
  acc1 = __builtin_amdgcn_wmma_f32_16x16x32_bf16(
      false, fa.v, false, fb1.v, (short)0, acc1, false, false);
}

// ---------------------------------------------------------------------------
// C[M,N] = act(A[M,K] @ B[K,N] + bias), bf16 in, f32 accum via WMMA.
// Block: 256 threads (8 waves). Tile: 128(M) x 32(N), K-step 32.
// TDM double buffering, K-loop unrolled x2 so each buffer index is a
// compile-time constant -> descriptor setup stays in SALU.
// ---------------------------------------------------------------------------
__global__ void __launch_bounds__(256)
wmma_gemm_bias(const __bf16* __restrict__ A, const __bf16* __restrict__ B,
               const float* __restrict__ bias, float* __restrict__ C,
               int M, int N, int K, int apply_tanh)
{
  __shared__ __align__(16) __bf16 As[2][BM * BK];  // 2 x 8 KB
  __shared__ __align__(16) __bf16 Bs[2][BK * BN];  // 2 x 2 KB

  const int tid  = threadIdx.x;
  const int wave = tid >> 5;
  const int lane = tid & 31;
  const int half = lane >> 4;
  const int l16  = lane & 15;

  const int n0 = blockIdx.x * BN;
  const int m0 = blockIdx.y * BM;

  v8f acc0 = {0.f, 0.f, 0.f, 0.f, 0.f, 0.f, 0.f, 0.f};
  v8f acc1 = {0.f, 0.f, 0.f, 0.f, 0.f, 0.f, 0.f, 0.f};

  // Prologue: DMA first K-slice into buffer 0.
  if (wave == 0) {
    tdm_load_2d_bf16(lds_off(&As[0][0]), A + (size_t)m0 * K, K, BK, BM);
    tdm_load_2d_bf16(lds_off(&Bs[0][0]), B + n0,             N, BN, BK);
  }

  for (int k0 = 0; k0 < K; k0 += 2 * BK) {
    // ---- phase 0: consume buf 0, prefetch k0+BK into buf 1 ----
    if (wave == 0) __builtin_amdgcn_s_wait_tensorcnt(0);
    __syncthreads();
    if (wave == 0 && k0 + BK < K) {
      tdm_load_2d_bf16(lds_off(&As[1][0]),
                       A + (size_t)m0 * K + (k0 + BK), K, BK, BM);
      tdm_load_2d_bf16(lds_off(&Bs[1][0]),
                       B + (size_t)(k0 + BK) * N + n0, N, BN, BK);
    }
    mma_step(&As[0][0], &Bs[0][0], wave, half, l16, acc0, acc1);
    __syncthreads();

    // ---- phase 1: consume buf 1, prefetch k0+2*BK into buf 0 ----
    if (k0 + BK < K) {
      if (wave == 0) __builtin_amdgcn_s_wait_tensorcnt(0);
      __syncthreads();
      if (wave == 0 && k0 + 2 * BK < K) {
        tdm_load_2d_bf16(lds_off(&As[0][0]),
                         A + (size_t)m0 * K + (k0 + 2 * BK), K, BK, BM);
        tdm_load_2d_bf16(lds_off(&Bs[0][0]),
                         B + (size_t)(k0 + 2 * BK) * N + n0, N, BN, BK);
      }
      mma_step(&As[1][0], &Bs[1][0], wave, half, l16, acc0, acc1);
      __syncthreads();
    }
  }

  // Epilogue: C layout VGPR r -> M = r + 8*half, lane -> N.
  const int mbase = m0 + wave * 16 + half * 8;
  const int colA = n0 + l16;
  const int colB = n0 + 16 + l16;
  const float bvA = bias[colA];
  const float bvB = bias[colB];
#pragma unroll
  for (int r = 0; r < 8; ++r) {
    float xa = acc0[r] + bvA;
    float xb = acc1[r] + bvB;
    if (apply_tanh) { xa = tanhf(xa); xb = tanhf(xb); }
    C[(size_t)(mbase + r) * N + colA] = xa;
    C[(size_t)(mbase + r) * N + colB] = xb;
  }
}

// ---------------------------------------------------------------------------
// Persistent recurrence: h_t = tanh(uX_t + h_{t-1} @ hW + hB).
// 64 WGs x 256 threads; WG g owns columns [g*32, g*32+32) of hW, cached in
// LDS as bf16 (128 KB of CDNA5's 320 KB WGP LDS). One grid barrier / step.
// Writes h_t as bf16 into H for the deferred output-projection GEMM.
// ---------------------------------------------------------------------------
__global__ void __launch_bounds__(256)
rnn_recurrence(const float* __restrict__ uX,      // [T, N_NODES]
               const __bf16* __restrict__ hW,     // [N_NODES, N_NODES]
               const float* __restrict__ hB,      // [N_NODES]
               float* __restrict__ hbuf,          // [2, N_NODES] ping-pong
               __bf16* __restrict__ H,            // [T, N_NODES]
               volatile int* __restrict__ bar)    // [T] step barriers
{
  extern __shared__ unsigned char smem[];
  __bf16* wl = (__bf16*)smem;                                   // [2048][32]
  float*  hl = (float*)(smem + (size_t)N_NODES * REC_COLS * 2); // [2048]

  const int g   = blockIdx.x;
  const int tid = threadIdx.x;

  // Cache this WG's hW column slice in LDS once (row-major, col fastest).
  for (int idx = tid; idx < N_NODES * REC_COLS; idx += 256) {
    int row = idx / REC_COLS, col = idx & (REC_COLS - 1);
    wl[idx] = hW[(size_t)row * N_NODES + g * REC_COLS + col];
  }
  __syncthreads();

  const int j    = tid >> 3;        // 0..31: column within the slice
  const int part = tid & 7;         // 8-way K split of the dot product
  const int colg = g * REC_COLS + j;
  const float bj = hB[colg];
  const int i0 = part * (N_NODES / 8);

  for (int t = 0; t < T_STEPS; ++t) {
    __threadfence();
    // stage h_{t-1} into LDS (8 KB)
    const float* hcur = hbuf + (size_t)(t & 1) * N_NODES;
    for (int i = tid; i < N_NODES; i += 256) hl[i] = hcur[i];
    // warm next step's uX slice into near caches while we compute
    if (t + 1 < T_STEPS)
      __builtin_prefetch(&uX[(size_t)(t + 1) * N_NODES + colg], 0, 3);
    __syncthreads();

    float acc = 0.f;
#pragma unroll 8
    for (int i = i0; i < i0 + N_NODES / 8; ++i)
      acc += hl[i] * (float)wl[i * REC_COLS + j];

    // reduce the 8 partial sums (lanes j*8+part live in one wave32)
    acc += __shfl_xor(acc, 1, 32);
    acc += __shfl_xor(acc, 2, 32);
    acc += __shfl_xor(acc, 4, 32);

    if (part == 0) {
      float hn = tanhf(uX[(size_t)t * N_NODES + colg] + bj + acc);
      hbuf[(size_t)((t + 1) & 1) * N_NODES + colg] = hn;
      H[(size_t)t * N_NODES + colg] = (__bf16)hn;
    }

    // device-wide barrier for step t
    __threadfence();
    __syncthreads();
    if (tid == 0) {
      atomicAdd((int*)&bar[t], 1);
      while (bar[t] < REC_WGS) __builtin_amdgcn_s_sleep(1);
    }
    __syncthreads();
  }
}

// ---------------------------------------------------------------------------
extern "C" void kernel_launch(void* const* d_in, const int* in_sizes, int n_in,
                              void* d_out, int out_size, void* d_ws,
                              size_t ws_size, hipStream_t stream)
{
  const float* x  = (const float*)d_in[0];  // [T, D_IN]
  const float* xW = (const float*)d_in[1];  // [D_IN, N_NODES]
  const float* xB = (const float*)d_in[2];  // [N_NODES]
  const float* hW = (const float*)d_in[3];  // [N_NODES, N_NODES]
  const float* hB = (const float*)d_in[4];  // [N_NODES]
  const float* yW = (const float*)d_in[5];  // [N_NODES, N_OUT]
  const float* yB = (const float*)d_in[6];  // [N_OUT]
  float* out = (float*)d_out;               // [T, N_OUT]
  (void)in_sizes; (void)n_in; (void)out_size; (void)ws_size;

  unsigned char* ws = (unsigned char*)d_ws;
  size_t off = 0;
  auto bump = [&](size_t bytes) -> void* {
    void* p = ws + off;
    off = (off + bytes + 255) & ~(size_t)255;
    return p;
  };
  __bf16* xbf  = (__bf16*)bump((size_t)T_STEPS * D_IN    * 2);  // 32 MB
  __bf16* xWbf = (__bf16*)bump((size_t)D_IN    * N_NODES * 2);  //  8 MB
  __bf16* hWbf = (__bf16*)bump((size_t)N_NODES * N_NODES * 2);  //  8 MB
  __bf16* yWbf = (__bf16*)bump((size_t)N_NODES * N_OUT   * 2);  //  8 MB
  float*  uX   = (float*) bump((size_t)T_STEPS * N_NODES * 4);  // 64 MB
  __bf16* H    = (__bf16*)bump((size_t)T_STEPS * N_NODES * 2);  // 32 MB
  float*  hbuf = (float*) bump((size_t)2 * N_NODES * 4);
  int*    bar  = (int*)   bump((size_t)T_STEPS * 4);

  // precision casts + state init
  cast_f32_bf16<<<512, 256, 0, stream>>>(x,  xbf,  T_STEPS * D_IN);
  cast_f32_bf16<<<512, 256, 0, stream>>>(xW, xWbf, D_IN * N_NODES);
  cast_f32_bf16<<<512, 256, 0, stream>>>(hW, hWbf, N_NODES * N_NODES);
  cast_f32_bf16<<<512, 256, 0, stream>>>(yW, yWbf, N_NODES * N_OUT);
  init_state<<<(T_STEPS + 255) / 256, 256, 0, stream>>>(hbuf, bar);

  // uX = x @ xW + xB   (dense WMMA GEMM, hoisted out of the scan)
  dim3 g1(N_NODES / BN, T_STEPS / BM);
  wmma_gemm_bias<<<g1, 256, 0, stream>>>(xbf, xWbf, xB, uX,
                                         T_STEPS, N_NODES, D_IN, 0);

  // sequential recurrence with LDS-resident hW
  size_t smem = (size_t)N_NODES * REC_COLS * sizeof(__bf16) +
                (size_t)N_NODES * sizeof(float);  // 136 KB
  (void)hipFuncSetAttribute(reinterpret_cast<const void*>(rnn_recurrence),
                            hipFuncAttributeMaxDynamicSharedMemorySize,
                            (int)smem);
  rnn_recurrence<<<REC_WGS, 256, smem, stream>>>(uX, hWbf, hB, hbuf, H, bar);

  // out = tanh(H @ yW + yB)   (second dense WMMA GEMM, hoisted out of scan)
  dim3 g2(N_OUT / BN, T_STEPS / BM);
  wmma_gemm_bias<<<g2, 256, 0, stream>>>(H, yWbf, yB, out,
                                         T_STEPS, N_OUT, N_NODES, 1);
}
